// Attention_50216757625003
// MI455X (gfx1250) — matrix operand order repaired and checked by reading, not compiled, not run on hardware
//
#include <hip/hip_runtime.h>

// ---------------------------------------------------------------------------
// Types for CDNA5 WMMA (wave32): bf16 16x16x32, f32 accumulate.
// ---------------------------------------------------------------------------
typedef __attribute__((ext_vector_type(16))) __bf16 bf16x16;
typedef __attribute__((ext_vector_type(8)))  __bf16 bf16x8;
typedef __attribute__((ext_vector_type(8)))  float  v8f;
typedef __attribute__((ext_vector_type(4)))  float  v4f;

union U16 { bf16x16 v; bf16x8 h[2]; };

__device__ __forceinline__ v8f wmma_bf16(bf16x16 a, bf16x16 b, v8f c) {
  return __builtin_amdgcn_wmma_f32_16x16x32_bf16(false, a, false, b, (short)0, c,
                                                 false, false);
}

// LDS byte offset of a generic pointer to __shared__ (low 32 bits of the
// generic address hold the wave-relative LDS offset).
__device__ __forceinline__ unsigned lds_off(const void* p) {
  return (unsigned)(uintptr_t)p;
}

// Async copy of 32 contiguous bytes global -> LDS (CDNA5, ASYNCcnt-tracked).
__device__ __forceinline__ void async_copy32(unsigned lds, const void* g) {
  asm volatile("global_load_async_to_lds_b128 %0, %1, off\n\t"
               "global_load_async_to_lds_b128 %0, %1, off offset:16"
               :: "v"(lds), "v"(g) : "memory");
}
// Async copy of 64 contiguous bytes global -> LDS.
__device__ __forceinline__ void async_copy64(unsigned lds, const void* g) {
  asm volatile("global_load_async_to_lds_b128 %0, %1, off\n\t"
               "global_load_async_to_lds_b128 %0, %1, off offset:16\n\t"
               "global_load_async_to_lds_b128 %0, %1, off offset:32\n\t"
               "global_load_async_to_lds_b128 %0, %1, off offset:48"
               :: "v"(lds), "v"(g) : "memory");
}
__device__ __forceinline__ void wait_async0() {
  asm volatile("s_wait_asynccnt 0x0" ::: "memory");
}
__device__ __forceinline__ void wait_ds0() {
  asm volatile("s_wait_dscnt 0x0" ::: "memory");
}

// ---------------------------------------------------------------------------
// Problem constants (from reference setup_inputs).
// ---------------------------------------------------------------------------
#define BB   2
#define SS   1024
#define DD   4096
#define HH   32
#define KVH  8
#define HD   128
#define AA   10
#define NREP 4
#define SKEYS 1034          /* A + S rows per (b, kv) in K/V buffers */

// ---------------------------------------------------------------------------
// Vectorized f32 -> bf16 converts (8 elems/thread; all sizes divisible by 8).
// ---------------------------------------------------------------------------
__global__ __launch_bounds__(256) void cvt_bf16x8(const float* __restrict__ s,
                                                  __bf16* __restrict__ d,
                                                  int n8) {
  int i = blockIdx.x * 256 + threadIdx.x;
  if (i >= n8) return;
  const v4f* sp = (const v4f*)(s + (size_t)i * 8);
  v4f a = sp[0], b = sp[1];
  bf16x8 o;
#pragma unroll
  for (int j = 0; j < 4; ++j) {
    o[j] = (__bf16)a[j];
    o[4 + j] = (__bf16)b[j];
  }
  *(bf16x8*)(d + (size_t)i * 8) = o;
}

__global__ __launch_bounds__(256) void cvt_bf16x8_pad(
    const float* __restrict__ s, __bf16* __restrict__ d, int nsrc8, int ntot8) {
  int i = blockIdx.x * 256 + threadIdx.x;
  if (i >= ntot8) return;
  bf16x8 o = {};
  if (i < nsrc8) {
    const v4f* sp = (const v4f*)(s + (size_t)i * 8);
    v4f a = sp[0], b = sp[1];
#pragma unroll
    for (int j = 0; j < 4; ++j) {
      o[j] = (__bf16)a[j];
      o[4 + j] = (__bf16)b[j];
    }
  }
  *(bf16x8*)(d + (size_t)i * 8) = o;
}

// ---------------------------------------------------------------------------
// Tiled bf16 GEMM: C[M,N](f32) = A[M,K](bf16, row-major) * B[K,N](bf16, rm).
// Block tile 128x128, 8 waves (2x4), per-wave 4x2 WMMA tiles, K-step 32.
// A tile staged with async global->LDS; B tile staged transposed via VGPRs.
// N, K must be multiples of 128 / 32 (true at all call sites); M guarded.
// ---------------------------------------------------------------------------
#define GA_STR 40   /* LDS row stride (elems) for A tile: conflict-free */
#define GB_STR 40   /* LDS row stride (elems) for transposed B tile      */

__global__ __launch_bounds__(256) void gemm_bf16(const __bf16* __restrict__ A,
                                                 const __bf16* __restrict__ B,
                                                 float* __restrict__ C,
                                                 int M, int N, int K) {
  __shared__ __bf16 sA[128 * GA_STR];
  __shared__ __bf16 sB[128 * GB_STR];   // transposed: sB[n][k]
  const int tid = threadIdx.x;
  const int lane = tid & 31, wave = tid >> 5;
  const int lane16 = lane & 15, hf = lane >> 4;
  const int wm = wave >> 2, wn = wave & 3;          // 2 x 4 wave grid
  const int row0 = blockIdx.y * 128, col0 = blockIdx.x * 128;

  v8f acc[4][2] = {};

  const int lrow = tid >> 1;          // 0..127 (A tile row)
  const int lk   = (tid & 1) * 16;    // 0 or 16
  const int bk   = tid >> 3;          // 0..31  (B tile k row)
  const int bn16 = (tid & 7) * 16;    // 0..112

  const int grow = row0 + lrow;
  const unsigned ldsA = lds_off(&sA[lrow * GA_STR + lk]);

  for (int k0 = 0; k0 < K; k0 += 32) {
    __syncthreads();
    {
      if (grow < M) {
        async_copy32(ldsA, A + (size_t)grow * K + k0 + lk);
      } else {
        bf16x8 z = {};
        ((bf16x8*)&sA[lrow * GA_STR + lk])[0] = z;
        ((bf16x8*)&sA[lrow * GA_STR + lk])[1] = z;
      }
      const __bf16* bsrc = B + (size_t)(k0 + bk) * N + col0 + bn16;
      bf16x8 b0 = ((const bf16x8*)bsrc)[0];
      bf16x8 b1 = ((const bf16x8*)bsrc)[1];
#pragma unroll
      for (int j = 0; j < 8; ++j) {
        sB[(bn16 + j) * GB_STR + bk]     = b0[j];
        sB[(bn16 + 8 + j) * GB_STR + bk] = b1[j];
      }
    }
    wait_async0();
    __syncthreads();

    bf16x16 afr[4], bfr[2];
#pragma unroll
    for (int i = 0; i < 4; ++i) {            // A frags: ISA 16-bit A layout
      U16 u;
      const __bf16* p = &sA[(wm * 64 + i * 16 + lane16) * GA_STR + hf * 8];
      u.h[0] = *(const bf16x8*)p;            // K 0..7   (or 8..15)
      u.h[1] = *(const bf16x8*)(p + 16);     // K 16..23 (or 24..31)
      afr[i] = u.v;
    }
#pragma unroll
    for (int j = 0; j < 2; ++j) {            // B frags: same K order as A
      U16 u;
      const __bf16* p = &sB[(wn * 32 + j * 16 + lane16) * GB_STR + hf * 8];
      u.h[0] = *(const bf16x8*)p;
      u.h[1] = *(const bf16x8*)(p + 16);
      bfr[j] = u.v;
    }
#pragma unroll
    for (int i = 0; i < 4; ++i)
#pragma unroll
      for (int j = 0; j < 2; ++j)
        acc[i][j] = wmma_bf16(afr[i], bfr[j], acc[i][j]);
  }

#pragma unroll
  for (int i = 0; i < 4; ++i)
#pragma unroll
    for (int j = 0; j < 2; ++j)
#pragma unroll
      for (int r = 0; r < 8; ++r) {          // C layout: row r + 8*half
        int gr = row0 + wm * 64 + i * 16 + r + hf * 8;
        int gc = col0 + wn * 32 + j * 16 + lane16;
        if (gr < M) C[(size_t)gr * N + gc] = acc[i][j][r];
      }
}

// ---------------------------------------------------------------------------
// RoPE on Q (interleaved pairs), write bf16 in [B][H][S][hd] layout.
// ---------------------------------------------------------------------------
__global__ __launch_bounds__(256) void rope_q(const float* __restrict__ qf,
                                              const float* __restrict__ cs,
                                              const float* __restrict__ sn,
                                              __bf16* __restrict__ qb) {
  int idx = blockIdx.x * 256 + threadIdx.x;   // B*S*H*64 pairs
  if (idx >= BB * SS * HH * 64) return;
  int i = idx & 63;
  int h = (idx >> 6) & 31;
  int s = (idx >> 11) & 1023;
  int b = idx >> 21;
  const float* src = qf + ((size_t)(b * SS + s)) * (HH * HD) + h * HD + 2 * i;
  float x0 = src[0], x1 = src[1];
  float c = cs[s * 64 + i], si = sn[s * 64 + i];
  __bf16* dst = qb + (((size_t)(b * HH + h)) * SS + s) * HD + 2 * i;
  dst[0] = (__bf16)(x0 * c - x1 * si);
  dst[1] = (__bf16)(x0 * si + x1 * c);
}

// RoPE on K, write into key rows [A + s] of [B][KV][SKEYS][hd].
__global__ __launch_bounds__(256) void rope_k(const float* __restrict__ kf,
                                              const float* __restrict__ cs,
                                              const float* __restrict__ sn,
                                              __bf16* __restrict__ kbuf) {
  int idx = blockIdx.x * 256 + threadIdx.x;   // B*S*KV*64 pairs
  if (idx >= BB * SS * KVH * 64) return;
  int i  = idx & 63;
  int kv = (idx >> 6) & 7;
  int s  = (idx >> 9) & 1023;
  int b  = idx >> 19;
  const float* src = kf + ((size_t)(b * SS + s)) * (KVH * HD) + kv * HD + 2 * i;
  float x0 = src[0], x1 = src[1];
  float c = cs[s * 64 + i], si = sn[s * 64 + i];
  __bf16* dst = kbuf + (((size_t)(b * KVH + kv)) * SKEYS + AA + s) * HD + 2 * i;
  dst[0] = (__bf16)(x0 * c - x1 * si);
  dst[1] = (__bf16)(x0 * si + x1 * c);
}

// V convert into value rows [A + s] of [B][KV][SKEYS][hd].
__global__ __launch_bounds__(256) void conv_v(const float* __restrict__ vf,
                                              __bf16* __restrict__ vbuf) {
  int idx = blockIdx.x * 256 + threadIdx.x;   // B*S*KV*128 elems
  if (idx >= BB * SS * KVH * HD) return;
  int d  = idx & 127;
  int kv = (idx >> 7) & 7;
  int s  = (idx >> 10) & 1023;
  int b  = idx >> 20;
  vbuf[(((size_t)(b * KVH + kv)) * SKEYS + AA + s) * HD + d] =
      (__bf16)vf[((size_t)(b * SS + s)) * (KVH * HD) + kv * HD + d];
}

// Adapter K/V (no RoPE) into rows 0..A-1, duplicated for both batches.
__global__ __launch_bounds__(256) void adapter_kv(const float* __restrict__ akf,
                                                  const float* __restrict__ avf,
                                                  __bf16* __restrict__ kbuf,
                                                  __bf16* __restrict__ vbuf) {
  int idx = blockIdx.x * 256 + threadIdx.x;   // B*KV*A*128
  if (idx >= BB * KVH * AA * HD) return;
  int d = idx & 127;
  int rest = idx >> 7;
  int j = rest % AA;
  int t = rest / AA;          // 0..15
  int kv = t & 7;
  int b  = t >> 3;
  size_t dst = (((size_t)(b * KVH + kv)) * SKEYS + j) * HD + d;
  size_t src = (size_t)j * (KVH * HD) + kv * HD + d;
  kbuf[dst] = (__bf16)akf[src];
  vbuf[dst] = (__bf16)avf[src];
}

// ---------------------------------------------------------------------------
// Flash attention with fused adapter block.
// grid = (B*H, S/128); block = 256 (8 waves, 16 q-rows each).
// ---------------------------------------------------------------------------
#define SKSTR 136   /* K tile LDS row stride (elems): conflict-free B-frags */
#define SVSTR 72    /* V^T tile LDS row stride */
#define PBSTR 72    /* per-wave P staging stride */

__global__ __launch_bounds__(256) void flash_kernel(
    const __bf16* __restrict__ qb, const __bf16* __restrict__ kbuf,
    const __bf16* __restrict__ vbuf, const float* __restrict__ gate,
    __bf16* __restrict__ attn) {
  __shared__ __bf16 sK[64 * SKSTR];
  __shared__ __bf16 sVt[128 * SVSTR];          // sVt[d][key]
  __shared__ __bf16 sP[8 * 16 * PBSTR];

  const int bh = blockIdx.x;
  const int b = bh / HH, h = bh % HH, kv = h / NREP;
  const int qr0 = blockIdx.y * 128;
  const int tid = threadIdx.x;
  const int wave = tid >> 5, lane = tid & 31;
  const int lane16 = lane & 15, hf = lane >> 4;
  const float scale = 0.088388347648318447f;   // 1/sqrt(128)

  const __bf16* kbase = kbuf + (((size_t)b * KVH + kv) * SKEYS) * HD;
  const __bf16* vbase = vbuf + (((size_t)b * KVH + kv) * SKEYS) * HD;

  // Q fragments for this wave's 16 rows, all 4 K-steps of hd=128.
  bf16x16 qfr[4];
  {
    int qr = qr0 + wave * 16 + lane16;
    const __bf16* qp = qb + ((((size_t)b * HH + h) * SS) + qr) * HD;
    int kh = hf * 8;
#pragma unroll
    for (int kk = 0; kk < 4; ++kk) {
      U16 u;
      u.h[0] = *(const bf16x8*)(qp + kk * 32 + kh);
      u.h[1] = *(const bf16x8*)(qp + kk * 32 + kh + 16);
      qfr[kk] = u.v;
    }
  }

  float m_i[8], l_i[8];
  v8f acc[8] = {};
#pragma unroll
  for (int r = 0; r < 8; ++r) { m_i[r] = -3.0e8f; l_i[r] = 0.0f; }

  const int nkb = qr0 / 64 + 2;       // key blocks of 64 covering causal range
  const int str = tid >> 2;           // staging row 0..63
  const int stc = (tid & 3) * 32;     // staging col 0,32,64,96
  const unsigned ldsK = lds_off(&sK[str * SKSTR + stc]);

  for (int kb = 0; kb < nkb; ++kb) {
    __syncthreads();
    {   // stage K block via async DMA; V block transposed via VGPRs
      async_copy64(ldsK, kbase + ((size_t)(AA + kb * 64 + str)) * HD + stc);
      const __bf16* vsrc = vbase + ((size_t)(AA + kb * 64 + str)) * HD + stc;
      bf16x8 vv[4];
#pragma unroll
      for (int q = 0; q < 4; ++q) vv[q] = ((const bf16x8*)vsrc)[q];
#pragma unroll
      for (int q = 0; q < 4; ++q)
#pragma unroll
        for (int j = 0; j < 8; ++j)
          sVt[(stc + q * 8 + j) * SVSTR + str] = vv[q][j];
    }
    wait_async0();
    __syncthreads();

    // scores: 4 n-tiles x 4 k-steps
    v8f sc[4];
#pragma unroll
    for (int nt = 0; nt < 4; ++nt) {
      v8f cc = {};
#pragma unroll
      for (int kk = 0; kk < 4; ++kk) {
        U16 u;
        const __bf16* bp = &sK[(nt * 16 + lane16) * SKSTR + kk * 32 + hf * 8];
        u.h[0] = *(const bf16x8*)bp;
        u.h[1] = *(const bf16x8*)(bp + 16);
        cc = wmma_bf16(qfr[kk], u.v, cc);
      }
      sc[nt] = cc;
    }

    // online softmax per row (rows r + 8*hf), reductions over 16-lane halves
    float fsc[8];
#pragma unroll
    for (int r = 0; r < 8; ++r) {
      int qr = qr0 + wave * 16 + r + hf * 8;
      float mx = -3.0e8f;
#pragma unroll
      for (int nt = 0; nt < 4; ++nt) {
        int sidx = kb * 64 + nt * 16 + lane16;
        float s = (sidx <= qr) ? sc[nt][r] * scale : -1.0e9f;
        sc[nt][r] = s;
        mx = fmaxf(mx, s);
      }
#pragma unroll
      for (int off = 1; off < 16; off <<= 1)
        mx = fmaxf(mx, __shfl_xor(mx, off, 32));
      float mnew = fmaxf(m_i[r], mx);
      fsc[r] = __expf(m_i[r] - mnew);
      m_i[r] = mnew;
      float rs = 0.0f;
#pragma unroll
      for (int nt = 0; nt < 4; ++nt) {
        float p = __expf(sc[nt][r] - mnew);
        sc[nt][r] = p;
        rs += p;
      }
#pragma unroll
      for (int off = 1; off < 16; off <<= 1)
        rs += __shfl_xor(rs, off, 32);
      l_i[r] = l_i[r] * fsc[r] + rs;
    }

    // rescale accumulators, stage P (C layout -> LDS -> A layout)
#pragma unroll
    for (int hn = 0; hn < 8; ++hn)
#pragma unroll
      for (int r = 0; r < 8; ++r) acc[hn][r] *= fsc[r];

    __bf16* pw = &sP[wave * 16 * PBSTR];
#pragma unroll
    for (int nt = 0; nt < 4; ++nt)
#pragma unroll
      for (int r = 0; r < 8; ++r)
        pw[(r + hf * 8) * PBSTR + nt * 16 + lane16] = (__bf16)sc[nt][r];
    wait_ds0();

#pragma unroll
    for (int ks = 0; ks < 2; ++ks) {
      U16 ua;
      const __bf16* ap = &pw[lane16 * PBSTR + ks * 32 + hf * 8];
      ua.h[0] = *(const bf16x8*)ap;
      ua.h[1] = *(const bf16x8*)(ap + 16);
#pragma unroll
      for (int hn = 0; hn < 8; ++hn) {
        U16 ub;
        const __bf16* bp = &sVt[(hn * 16 + lane16) * SVSTR + ks * 32 + hf * 8];
        ub.h[0] = *(const bf16x8*)bp;
        ub.h[1] = *(const bf16x8*)(bp + 16);
        acc[hn] = wmma_bf16(ua.v, ub.v, acc[hn]);
      }
    }
  }

  // -------- adapter block (separate softmax * tanh(gate), fused via WMMA-C) --
  __syncthreads();
  if (tid < 64) {       // adapter K rows 0..15 via async DMA
    int r = tid >> 2, c = (tid & 3) * 32;
    async_copy64(lds_off(&sK[r * SKSTR + c]), kbase + (size_t)r * HD + c);
  }
  if (tid < 128) {      // V rows 0..31 transposed (rows 16..31 get zero P)
    int r = tid >> 2, c = (tid & 3) * 32;
    const __bf16* vsrc = vbase + (size_t)r * HD + c;
    bf16x8 vv[4];
#pragma unroll
    for (int q = 0; q < 4; ++q) vv[q] = ((const bf16x8*)vsrc)[q];
#pragma unroll
    for (int q = 0; q < 4; ++q)
#pragma unroll
      for (int j = 0; j < 8; ++j)
        sVt[(c + q * 8 + j) * SVSTR + r] = vv[q][j];
  }
  wait_async0();
  __syncthreads();

  v8f ca = {};
#pragma unroll
  for (int kk = 0; kk < 4; ++kk) {
    U16 u;
    const __bf16* bp = &sK[lane16 * SKSTR + kk * 32 + hf * 8];
    u.h[0] = *(const bf16x8*)bp;
    u.h[1] = *(const bf16x8*)(bp + 16);
    ca = wmma_bf16(qfr[kk], u.v, ca);
  }

  // finalize causal part: divide by l
#pragma unroll
  for (int r = 0; r < 8; ++r) {
    float inv = 1.0f / l_i[r];
#pragma unroll
    for (int hn = 0; hn < 8; ++hn) acc[hn][r] *= inv;
  }

  float tg = tanhf(gate[h]);
  __bf16* pw = &sP[wave * 16 * PBSTR];
#pragma unroll
  for (int r = 0; r < 8; ++r) {
    float s = (lane16 < AA) ? ca[r] * scale : -1.0e9f;
    float mx = s;
#pragma unroll
    for (int off = 1; off < 16; off <<= 1)
      mx = fmaxf(mx, __shfl_xor(mx, off, 32));
    float p = __expf(s - mx);
    float rs = p;
#pragma unroll
    for (int off = 1; off < 16; off <<= 1)
      rs += __shfl_xor(rs, off, 32);
    float pn = p / rs * tg;
    int row = r + hf * 8;
    pw[row * PBSTR + lane16] = (__bf16)pn;
    pw[row * PBSTR + 16 + lane16] = (__bf16)0.0f;   // zero K-pad cols 16..31
  }
  wait_ds0();
  {
    U16 ua;
    const __bf16* ap = &pw[lane16 * PBSTR + hf * 8];
    ua.h[0] = *(const bf16x8*)ap;
    ua.h[1] = *(const bf16x8*)(ap + 16);
#pragma unroll
    for (int hn = 0; hn < 8; ++hn) {
      U16 ub;
      const __bf16* bp = &sVt[(hn * 16 + lane16) * SVSTR + hf * 8];
      ub.h[0] = *(const bf16x8*)bp;
      ub.h[1] = *(const bf16x8*)(bp + 16);
      acc[hn] = wmma_bf16(ua.v, ub.v, acc[hn]);   // out += P_adapter @ V_adapter
    }
  }

  // store bf16 attention output as [b*S + qr][h*128 + d]
#pragma unroll
  for (int hn = 0; hn < 8; ++hn)
#pragma unroll
    for (int r = 0; r < 8; ++r) {
      int qr = qr0 + wave * 16 + r + hf * 8;
      attn[(((size_t)b * SS + qr) * (HH * HD)) + h * HD + hn * 16 + lane16] =
          (__bf16)acc[hn][r];
    }
}

// ---------------------------------------------------------------------------
// Host-side orchestration.
// ---------------------------------------------------------------------------
extern "C" void kernel_launch(void* const* d_in, const int* in_sizes, int n_in,
                              void* d_out, int out_size, void* d_ws,
                              size_t ws_size, hipStream_t stream) {
  (void)in_sizes; (void)n_in; (void)out_size; (void)ws_size;
  const float* x       = (const float*)d_in[0];
  const float* wq      = (const float*)d_in[1];
  const float* wk      = (const float*)d_in[2];
  const float* wv      = (const float*)d_in[3];
  const float* wo      = (const float*)d_in[4];
  const float* adapter = (const float*)d_in[5];
  const float* gate    = (const float*)d_in[6];
  const float* fcos    = (const float*)d_in[7];
  const float* fsin    = (const float*)d_in[8];
  // d_in[9] (mask) unused: causal mask computed analytically.
  float* out = (float*)d_out;

  char* w = (char*)d_ws;
  auto carve = [&](size_t bytes) {
    char* p = w;
    w += (bytes + 255) & ~(size_t)255;
    return p;
  };
  const size_t nx  = (size_t)BB * SS * DD;         // 8388608
  const size_t nwq = (size_t)DD * DD;              // 16777216
  const size_t nwk = (size_t)DD * (KVH * HD);      // 4194304
  const size_t nkv = (size_t)BB * SS * KVH * HD;   // 2097152
  const size_t nkb = (size_t)BB * KVH * SKEYS * HD;

  __bf16* xb  = (__bf16*)carve(nx * 2);
  __bf16* wqb = (__bf16*)carve(nwq * 2);
  __bf16* wkb = (__bf16*)carve(nwk * 2);
  __bf16* wvb = (__bf16*)carve(nwk * 2);
  __bf16* wob = (__bf16*)carve(nwq * 2);
  __bf16* adb = (__bf16*)carve((size_t)16 * DD * 2);
  float*  qf  = (float*)carve(nx * 4);             // also reused for attn out
  float*  kf  = (float*)carve(nkv * 4);
  float*  vf  = (float*)carve(nkv * 4);
  float*  akf = (float*)carve((size_t)16 * KVH * HD * 4);
  float*  avf = (float*)carve((size_t)16 * KVH * HD * 4);
  __bf16* qb   = (__bf16*)carve(nx * 2);
  __bf16* kbuf = (__bf16*)carve(nkb * 2);
  __bf16* vbuf = (__bf16*)carve(nkb * 2);
  __bf16* at   = (__bf16*)qf;   // alias: qf dead after rope_q

  cvt_bf16x8<<<(int)(nx / 8 / 256), 256, 0, stream>>>(x, xb, (int)(nx / 8));
  cvt_bf16x8<<<(int)(nwq / 8 / 256), 256, 0, stream>>>(wq, wqb, (int)(nwq / 8));
  cvt_bf16x8<<<(int)(nwk / 8 / 256), 256, 0, stream>>>(wk, wkb, (int)(nwk / 8));
  cvt_bf16x8<<<(int)(nwk / 8 / 256), 256, 0, stream>>>(wv, wvb, (int)(nwk / 8));
  cvt_bf16x8<<<(int)(nwq / 8 / 256), 256, 0, stream>>>(wo, wob, (int)(nwq / 8));
  cvt_bf16x8_pad<<<(16 * DD / 8 + 255) / 256, 256, 0, stream>>>(
      adapter, adb, AA * DD / 8, 16 * DD / 8);

  gemm_bf16<<<dim3(32, 16), 256, 0, stream>>>(xb, wqb, qf, 2048, 4096, 4096);
  gemm_bf16<<<dim3(8, 16), 256, 0, stream>>>(xb, wkb, kf, 2048, 1024, 4096);
  gemm_bf16<<<dim3(8, 16), 256, 0, stream>>>(xb, wvb, vf, 2048, 1024, 4096);
  gemm_bf16<<<dim3(8, 1), 256, 0, stream>>>(adb, wkb, akf, 16, 1024, 4096);
  gemm_bf16<<<dim3(8, 1), 256, 0, stream>>>(adb, wvb, avf, 16, 1024, 4096);

  rope_q<<<(BB * SS * HH * 64) / 256, 256, 0, stream>>>(qf, fcos, fsin, qb);
  rope_k<<<(BB * SS * KVH * 64) / 256, 256, 0, stream>>>(kf, fcos, fsin, kbuf);
  conv_v<<<(BB * SS * KVH * HD) / 256, 256, 0, stream>>>(vf, vbuf);
  adapter_kv<<<(BB * KVH * AA * HD + 255) / 256, 256, 0, stream>>>(akf, avf,
                                                                   kbuf, vbuf);

  flash_kernel<<<dim3(BB * HH, SS / 128), 256, 0, stream>>>(qb, kbuf, vbuf,
                                                            gate, at);

  gemm_bf16<<<dim3(32, 16), 256, 0, stream>>>(at, wob, out, 2048, 4096, 4096);
}
